// SparseLinear_29695403884701
// MI455X (gfx1250) — matrix operand order repaired and checked by reading, not compiled, not run on hardware
//
#include <hip/hip_runtime.h>

// ---------------------------------------------------------------------------
// SparseLinear on MI455X (gfx1250):
//   out = X @ dense(W_coo)^T + bias,  X:[8192,4096] f32, W:4096x4096 @10%
// Densify W (fp32 atomics), split W and X into bf16 hi/lo pairs, then a
// wave32 WMMA GEMM computing C += Ah*Bh + Al*Bh + Ah*Bl (fp32-grade bf16x3)
// with v_wmma_f32_16x16x32_bf16, 64x64 register tile per wave.
//
// GEMM operand feed uses the CDNA5 async DMA path: each K-tile's unique
// bytes (W: 32KB hi+lo, A: 16KB hi+lo) are staged global->LDS with
// global_load_async_to_lds_b128 (ASYNCcnt), double-buffered so the DMA for
// tile k+1 overlaps the 48 WMMAs of tile k; waves read fragments from LDS
// (ds_load_b128). HBM roofline ~320MB (~14us @23.3TB/s); W(64MB) is
// L2-resident (192MB L2).
//
// ws layout (192 MiB):
//   [  0, 64MiB) : dense W fp32 (phases 1-3), then reused as A_hi bf16
//   [ 64, 96MiB) : W_hi bf16
//   [ 96,128MiB) : W_lo bf16
//   [128,192MiB) : A_lo bf16
// ---------------------------------------------------------------------------

typedef __attribute__((ext_vector_type(4)))  __bf16 v4bf;
typedef __attribute__((ext_vector_type(8)))  __bf16 v8bf;
typedef __attribute__((ext_vector_type(16))) __bf16 v16bf;
typedef __attribute__((ext_vector_type(8)))  float  v8f;

#define IN_F  4096
#define OUT_F 4096
#define N_TOK 8192

// ---- phase 1: zero the dense-W accumulator -------------------------------
__global__ void SparseLinear_zero_kernel(float4* __restrict__ p) {
  size_t i = (size_t)blockIdx.x * blockDim.x + threadIdx.x;
  p[i] = float4{0.f, 0.f, 0.f, 0.f};
}

// ---- phase 2: COO scatter with duplicate summation -----------------------
__global__ void SparseLinear_scatter_kernel(const int* __restrict__ rows,
                                            const int* __restrict__ cols,
                                            const float* __restrict__ vals,
                                            float* __restrict__ Wd, int nnz) {
  int i = blockIdx.x * blockDim.x + threadIdx.x;
  if (i < nnz) {
    atomicAdd(&Wd[(size_t)rows[i] * IN_F + cols[i]], vals[i]);
  }
}

// ---- phases 3/4: fp32 -> bf16 hi/lo split (4 elems / thread) -------------
__global__ void SparseLinear_split_kernel(const float* __restrict__ src,
                                          __bf16* __restrict__ hi,
                                          __bf16* __restrict__ lo) {
  size_t i = ((size_t)blockIdx.x * blockDim.x + threadIdx.x) * 4;
  float4 v = *(const float4*)(src + i);
  __bf16 h0 = (__bf16)v.x, h1 = (__bf16)v.y, h2 = (__bf16)v.z, h3 = (__bf16)v.w;
  v4bf H = {h0, h1, h2, h3};
  v4bf L = {(__bf16)(v.x - (float)h0), (__bf16)(v.y - (float)h1),
            (__bf16)(v.z - (float)h2), (__bf16)(v.w - (float)h3)};
  *(v4bf*)(hi + i) = H;
  *(v4bf*)(lo + i) = L;
}

// ---- phase 5: WMMA GEMM + bias -------------------------------------------
// Block: 256 threads = 8 waves (2 in M x 4 in N). Block tile 128x256.
// Wave tile 64x64 = 4x4 fragments of 16x16; K step 32.
// LDS (dynamic, 96KB): sW[2][2][256*32] (64KB) + sA[2][2][128*32] (32KB).
#define BM 128
#define BN 256
#define LDS_W_ELEMS (BN * 32)   // per (buf,prec): 256 rows x 32 K
#define LDS_A_ELEMS (BM * 32)   // per (buf,prec): 128 rows x 32 K
#define LDS_BYTES (2 * 2 * (LDS_W_ELEMS + LDS_A_ELEMS) * 2)

__device__ __forceinline__ void async_b128(unsigned lds_off,
                                           const __bf16* gptr) {
  asm volatile("global_load_async_to_lds_b128 %0, %1, off"
               :: "v"(lds_off), "v"((unsigned long long)(uintptr_t)gptr)
               : "memory");
}

__global__ __launch_bounds__(256) void
SparseLinear_gemm_kernel(const __bf16* __restrict__ Ah,
                         const __bf16* __restrict__ Al,
                         const __bf16* __restrict__ Wh,
                         const __bf16* __restrict__ Wl,
                         const float* __restrict__ bias,
                         float* __restrict__ out) {
  extern __shared__ __align__(128) char dynsmem[];
  __bf16* sW = (__bf16*)dynsmem;                                   // [2][2][BN*32]
  __bf16* sA = (__bf16*)(dynsmem + 2 * 2 * LDS_W_ELEMS * 2);       // [2][2][BM*32]

  const int tid  = threadIdx.x;
  const int lane = tid & 31;
  const int wave = tid >> 5;
  const int half = lane >> 4;   // lane-half selects K sub-range of fragments
  const int ml   = lane & 15;   // M (A rows) or N (B cols / C cols)

  const int bm0 = blockIdx.y * BM;
  const int bn0 = blockIdx.x * BN;
  const int wm  = (wave & 1) * 64;
  const int wn  = (wave >> 1) * 64;

  const int k0a = half * 16;  // A frag: lanes 0-15 K 0..15; lanes 16-31 K 16..31
  const int k0b = half * 16;  // B frag: lanes 0-15 K 0..15; lanes 16-31 K 16..31

  v8f c[4][4] = {};

  // --- async stage of one K-tile into buffer `buf` (12 b128 per thread) ---
  auto stage = [&](int buf, int kk) {
    // W: thread t owns row t of the 256-row tile; 4 x 16B chunks per prec
    const __bf16* gh = Wh + (size_t)(bn0 + tid) * IN_F + kk;
    const __bf16* gl = Wl + (size_t)(bn0 + tid) * IN_F + kk;
    unsigned lwh = (unsigned)(uintptr_t)(sW + (buf * 2 + 0) * LDS_W_ELEMS + tid * 32);
    unsigned lwl = (unsigned)(uintptr_t)(sW + (buf * 2 + 1) * LDS_W_ELEMS + tid * 32);
#pragma unroll
    for (int cc = 0; cc < 4; ++cc) {
      async_b128(lwh + cc * 16, gh + cc * 8);
      async_b128(lwl + cc * 16, gl + cc * 8);
    }
    // A: 2 threads per row; thread t owns 16-elem half-row (32B)
    const int ar = tid >> 1, ac = (tid & 1) * 16;
    const __bf16* gah = Ah + (size_t)(bm0 + ar) * IN_F + kk + ac;
    const __bf16* gal = Al + (size_t)(bm0 + ar) * IN_F + kk + ac;
    unsigned lah = (unsigned)(uintptr_t)(sA + (buf * 2 + 0) * LDS_A_ELEMS + ar * 32 + ac);
    unsigned lal = (unsigned)(uintptr_t)(sA + (buf * 2 + 1) * LDS_A_ELEMS + ar * 32 + ac);
#pragma unroll
    for (int cc = 0; cc < 2; ++cc) {
      async_b128(lah + cc * 16, gah + cc * 8);
      async_b128(lal + cc * 16, gal + cc * 8);
    }
  };

  stage(0, 0);

#pragma unroll 1
  for (int kk = 0; kk < IN_F; kk += 32) {
    const int buf = (kk >> 5) & 1;
    if (kk + 32 < IN_F) {
      stage(buf ^ 1, kk + 32);  // prefetch next tile; 12 async in flight
      asm volatile("s_wait_asynccnt 0xc" ::: "memory");  // current tile landed
    } else {
      asm volatile("s_wait_asynccnt 0x0" ::: "memory");
    }
    __syncthreads();

    const __bf16* swh = sW + (buf * 2 + 0) * LDS_W_ELEMS;
    const __bf16* swl = sW + (buf * 2 + 1) * LDS_W_ELEMS;
    const __bf16* sah = sA + (buf * 2 + 0) * LDS_A_ELEMS;
    const __bf16* sal = sA + (buf * 2 + 1) * LDS_A_ELEMS;

    v16bf ah[4], al[4];
#pragma unroll
    for (int fm = 0; fm < 4; ++fm) {
      const int rb = (wm + fm * 16 + ml) * 32 + k0a;
      v8bf a0 = *(const v8bf*)(sah + rb);
      v8bf a1 = *(const v8bf*)(sah + rb + 8);
      v8bf l0 = *(const v8bf*)(sal + rb);
      v8bf l1 = *(const v8bf*)(sal + rb + 8);
      ah[fm] = __builtin_shufflevector(a0, a1, 0,1,2,3,4,5,6,7,8,9,10,11,12,13,14,15);
      al[fm] = __builtin_shufflevector(l0, l1, 0,1,2,3,4,5,6,7,8,9,10,11,12,13,14,15);
    }
#pragma unroll
    for (int fn = 0; fn < 4; ++fn) {
      const int nb = (wn + fn * 16 + ml) * 32 + k0b;
      v16bf bh = *(const v16bf*)(swh + nb);
      v16bf bl = *(const v16bf*)(swl + nb);
#pragma unroll
      for (int fm = 0; fm < 4; ++fm) {
        c[fm][fn] = __builtin_amdgcn_wmma_f32_16x16x32_bf16(
            false, ah[fm], false, bh, (short)0, c[fm][fn], false, false);
        c[fm][fn] = __builtin_amdgcn_wmma_f32_16x16x32_bf16(
            false, al[fm], false, bh, (short)0, c[fm][fn], false, false);
        c[fm][fn] = __builtin_amdgcn_wmma_f32_16x16x32_bf16(
            false, ah[fm], false, bl, (short)0, c[fm][fn], false, false);
      }
    }
    __syncthreads();  // all reads of `buf` done before it is re-staged
  }

  // Epilogue: C layout = 8 VGPRs; lanes 0-15 rows M=r, lanes 16-31 rows M=8+r
#pragma unroll
  for (int fn = 0; fn < 4; ++fn) {
    const int n = bn0 + wn + fn * 16 + ml;
    const float bv = bias[n];
#pragma unroll
    for (int fm = 0; fm < 4; ++fm) {
      const int mbase = bm0 + wm + fm * 16 + half * 8;
#pragma unroll
      for (int r = 0; r < 8; ++r) {
        out[(size_t)(mbase + r) * OUT_F + n] = c[fm][fn][r] + bv;
      }
    }
  }
}

extern "C" void kernel_launch(void* const* d_in, const int* in_sizes, int n_in,
                              void* d_out, int out_size, void* d_ws, size_t ws_size,
                              hipStream_t stream) {
  const float* x    = (const float*)d_in[0];   // [8192, 4096] f32
  const int*   idx  = (const int*)d_in[1];     // [2, nnz] i32
  const float* vals = (const float*)d_in[2];   // [nnz] f32
  const float* bias = (const float*)d_in[3];   // [4096] f32
  float* out = (float*)d_out;
  const int nnz = in_sizes[2];

  char* ws = (char*)d_ws;
  float*  Wd  = (float*)ws;                             // 64 MiB (phases 1-3)
  __bf16* Wh  = (__bf16*)(ws + ((size_t)64 << 20));     // 32 MiB
  __bf16* Wl  = (__bf16*)(ws + ((size_t)96 << 20));     // 32 MiB
  __bf16* Ahi = (__bf16*)ws;                            // reuses Wd region
  __bf16* Alo = (__bf16*)(ws + ((size_t)128 << 20));    // 64 MiB

  // 1) zero dense W
  SparseLinear_zero_kernel<<<(OUT_F * IN_F / 4) / 256, 256, 0, stream>>>((float4*)Wd);
  // 2) COO scatter (duplicates sum, matching reference .at[].add)
  SparseLinear_scatter_kernel<<<(nnz + 255) / 256, 256, 0, stream>>>(
      idx, idx + nnz, vals, Wd, nnz);
  // 3) W -> bf16 hi/lo
  SparseLinear_split_kernel<<<(OUT_F * IN_F / 4) / 256, 256, 0, stream>>>(Wd, Wh, Wl);
  // 4) X -> bf16 hi/lo (overwrites Wd region; ordered after step 3 on stream)
  SparseLinear_split_kernel<<<(N_TOK * IN_F / 4) / 256, 256, 0, stream>>>(x, Ahi, Alo);
  // 5) GEMM + bias, async-LDS double-buffered
  dim3 grid(OUT_F / BN, N_TOK / BM);
  SparseLinear_gemm_kernel<<<grid, 256, LDS_BYTES, stream>>>(Ahi, Alo, Wh, Wl, bias, out);
}
